// FeatureGNNModel_549755814533
// MI455X (gfx1250) — matrix operand order repaired and checked by reading, not compiled, not run on hardware
//
#include <hip/hip_runtime.h>
#include <hip/hip_bf16.h>

#define HDIM   64
#define NUSERS 100000
#define NITEMS 50000
#define NNODES 150000
#define LNEPS  1e-5f
#define SLD    68   // padded f32 LDS row stride (bank-conflict avoidance)

typedef __attribute__((ext_vector_type(16))) __bf16 v16bf;
typedef __attribute__((ext_vector_type(8)))  float  v8f;

// ---- CDNA5 async global->LDS DMA (ASYNCcnt path, ISA 08_async_tensor §4) --
// LDS operand VGPR holds the LDS byte offset == low 32 bits of the generic
// pointer to a __shared__ object (flat LDS aperture truncation, ISA §10.2).
__device__ __forceinline__ void async_load_lds_b128(void* lds_dst,
                                                    const void* gsrc) {
  unsigned int lds = (unsigned int)(unsigned long long)lds_dst;
  asm volatile("global_load_async_to_lds_b128 %0, %1, off"
               :: "v"(lds), "v"(gsrc)
               : "memory");
}
__device__ __forceinline__ void wait_async_lds() {
  asm volatile("s_wait_asynccnt 0x0" ::: "memory");
}

// ---- WMMA fragment loaders (CDNA5 16x16x32 bf16 layouts, wave32) ----------

// A: 16x32 bf16 tile, row-major lda. lane m = lane&15, K-base = (lane>>4)*8;
// elems 0..7 -> K=kb..kb+7, elems 8..15 -> K=kb+16..kb+23
__device__ __forceinline__ v16bf load_a_frag(const __bf16* A, int lda,
                                             int lane, int k0) {
  int m  = lane & 15;
  int kb = k0 + ((lane >> 4) << 3);
  const __bf16* p = A + m * lda + kb;
  v16bf a;
#pragma unroll
  for (int i = 0; i < 8; ++i) a[i] = p[i];
#pragma unroll
  for (int i = 0; i < 8; ++i) a[i + 8] = p[i + 16];
  return a;
}

// Same A fragment, but from raw f32 staged in LDS (async DMA path):
// convert f32 -> bf16 in registers while loading.
__device__ __forceinline__ v16bf load_a_frag(const float* A, int lda,
                                             int lane, int k0) {
  int m  = lane & 15;
  int kb = k0 + ((lane >> 4) << 3);
  const float* p = A + m * lda + kb;
  v16bf a;
#pragma unroll
  for (int i = 0; i < 8; ++i) a[i] = (__bf16)p[i];
#pragma unroll
  for (int i = 0; i < 8; ++i) a[i + 8] = (__bf16)p[i + 16];
  return a;
}

// B: 32x16 (K x N) where B[k][n] = W[n][k], W row-major f32 [N][ldw].
// lane n = lane&15, kb = k0 + (lane>>4)*8; elems 0..7 -> K=kb..kb+7, 8..15 -> K=kb+16..kb+23.
__device__ __forceinline__ v16bf load_b_frag_w(const float* W, int ldw,
                                               int lane, int n0, int k0) {
  int n  = n0 + (lane & 15);
  int kb = k0 + ((lane >> 4) << 3);
  const float* p = W + (size_t)n * ldw + kb;
  v16bf b;
#pragma unroll
  for (int i = 0; i < 8; ++i) b[i] = (__bf16)p[i];
#pragma unroll
  for (int i = 0; i < 8; ++i) b[i + 8] = (__bf16)p[i + 16];
  return b;
}

// One wave computes a 16x16 f32 tile: out = A[16xK] * W^T + bias (opt relu),
// written to padded f32 LDS S[16][lds_]. AT is __bf16 or float (LDS staging).
template <int K, typename AT>
__device__ __forceinline__ void wave_gemm_bias_relu(const AT* A, int lda,
                                                    const float* W, int ldw,
                                                    const float* bias, float* S,
                                                    int lds_, int lane, int n0,
                                                    bool relu) {
  v8f c = {};
#pragma unroll
  for (int kt = 0; kt < K / 32; ++kt) {
    v16bf a = load_a_frag(A, lda, lane, kt * 32);
    v16bf b = load_b_frag_w(W, ldw, lane, n0, kt * 32);
    c = __builtin_amdgcn_wmma_f32_16x16x32_bf16(false, a, false, b,
                                                (short)0, c, false, false);
  }
  int   n  = n0 + (lane & 15);
  int   mb = (lane < 16) ? 0 : 8;   // C/D layout: vgpr r -> row mb+r
  float bv = bias[n];
#pragma unroll
  for (int r = 0; r < 8; ++r) {
    float v = c[r] + bv;
    if (relu) v = fmaxf(v, 0.0f);
    S[(size_t)(mb + r) * lds_ + n] = v;
  }
}

// Row LayerNorm stats over 64 cols: 8 threads/row, shfl-xor reduce in group of 8.
__device__ __forceinline__ float2 ln_stats(const float* S, int lds_, int tid) {
  int row = tid >> 3, g = tid & 7;
  const float* p = S + (size_t)row * lds_ + g * 8;
  float s = 0.f, s2 = 0.f;
#pragma unroll
  for (int i = 0; i < 8; ++i) { float v = p[i]; s += v; s2 += v * v; }
  s  += __shfl_xor(s, 4, 8);  s  += __shfl_xor(s, 2, 8);  s  += __shfl_xor(s, 1, 8);
  s2 += __shfl_xor(s2, 4, 8); s2 += __shfl_xor(s2, 2, 8); s2 += __shfl_xor(s2, 1, 8);
  float mean = s * (1.0f / 64.0f);
  float var  = s2 * (1.0f / 64.0f) - mean * mean;
  return make_float2(mean, rsqrtf(var + LNEPS));
}

// ---- Kernel 1: fused encoder: LN(relu(feat@encW^T+b)) -> concat(emb) ->
//      LN(relu(.@fusW^T+b)) -> x0 and acc. 16 rows/block, 4 waves. ----------
__global__ __launch_bounds__(128) void enc_fuse_kernel(
    const float* __restrict__ feat,
    const float* __restrict__ encW, const float* __restrict__ encB,
    const float* __restrict__ encG, const float* __restrict__ encBE,
    const float* __restrict__ emb,
    const float* __restrict__ fusW, const float* __restrict__ fusB,
    const float* __restrict__ fusG, const float* __restrict__ fusBE,
    float* __restrict__ x_out, float* __restrict__ acc_out) {
  __shared__ float  Xf[16][128];    // raw f32 feature tile (async DMA target)
  __shared__ __bf16 Abuf[16][128];  // stage-B input [h | emb] in bf16
  __shared__ float  Sbuf[16][SLD];
  int tid = threadIdx.x, lane = tid & 31, w = tid >> 5;
  int row0 = blockIdx.x * 16;

  // features tile [16x128] f32 -> LDS via async DMA (16B per thread per op)
#pragma unroll
  for (int i = 0; i < 4; ++i) {
    int e = (tid + i * 128) * 4;         // float index in [0, 2048)
    int r = e >> 7, c = e & 127;
    async_load_lds_b128(&Xf[r][c], &feat[(size_t)(row0 + r) * 128 + c]);
  }
  wait_async_lds();
  __syncthreads();
  wave_gemm_bias_relu<128>(&Xf[0][0], 128, encW, 128, encB,
                           &Sbuf[0][0], SLD, lane, w * 16, true);
  __syncthreads();
  // LayerNorm -> h (bf16) into Abuf[:,0:64]
  {
    float2 st = ln_stats(&Sbuf[0][0], SLD, tid);
    int row = tid >> 3, g = tid & 7;
#pragma unroll
    for (int i = 0; i < 8; ++i) {
      int c = g * 8 + i;
      float v = (Sbuf[row][c] - st.x) * st.y * encG[c] + encBE[c];
      Abuf[row][c] = (__bf16)v;
    }
  }
  // id-embedding tile into Abuf[:,64:128]
#pragma unroll
  for (int i = 0; i < 8; ++i) {
    int e = tid + i * 128;
    int r = e >> 6, c = e & 63;
    Abuf[r][64 + c] = (__bf16)emb[(size_t)(row0 + r) * 64 + c];
  }
  __syncthreads();
  wave_gemm_bias_relu<128>(&Abuf[0][0], 128, fusW, 128, fusB,
                           &Sbuf[0][0], SLD, lane, w * 16, true);
  __syncthreads();
  {
    float2 st = ln_stats(&Sbuf[0][0], SLD, tid);
    int row = tid >> 3, g = tid & 7;
#pragma unroll
    for (int i = 0; i < 8; ++i) {
      int c = g * 8 + i;
      float v = (Sbuf[row][c] - st.x) * st.y * fusG[c] + fusBE[c];
      size_t o = (size_t)(row0 + row) * 64 + c;
      x_out[o]   = v;
      acc_out[o] = v;
    }
  }
}

// ---- Kernel 2: GCN per-node linear y = x @ W^T + b (64->64) ---------------
__global__ __launch_bounds__(128) void gcn_linear_kernel(
    const float* __restrict__ x, const float* __restrict__ W,
    const float* __restrict__ bias, float* __restrict__ y) {
  __shared__ float Xf[16][64];      // raw f32 tile via async DMA
  int tid = threadIdx.x, lane = tid & 31, w = tid >> 5;
  int row0 = blockIdx.x * 16;
#pragma unroll
  for (int i = 0; i < 2; ++i) {
    int e = (tid + i * 128) * 4;         // float index in [0, 1024)
    int r = e >> 6, c = e & 63;
    async_load_lds_b128(&Xf[r][c], &x[(size_t)(row0 + r) * 64 + c]);
  }
  wait_async_lds();
  __syncthreads();
  v8f cacc = {};
#pragma unroll
  for (int kt = 0; kt < 2; ++kt) {
    v16bf a = load_a_frag(&Xf[0][0], 64, lane, kt * 32);
    v16bf b = load_b_frag_w(W, 64, lane, w * 16, kt * 32);
    cacc = __builtin_amdgcn_wmma_f32_16x16x32_bf16(false, a, false, b,
                                                   (short)0, cacc, false, false);
  }
  int   n  = w * 16 + (lane & 15);
  int   mb = (lane < 16) ? 0 : 8;
  float bv = bias[n];
#pragma unroll
  for (int r = 0; r < 8; ++r)
    y[(size_t)(row0 + mb + r) * 64 + n] = cacc[r] + bv;
}

// ---- Kernel 3: zero scratch (b128 stores) ---------------------------------
__global__ void zero_kernel(float4* __restrict__ p, int n4) {
  int i = blockIdx.x * blockDim.x + threadIdx.x;
  int stride = gridDim.x * blockDim.x;
  float4 z = make_float4(0.f, 0.f, 0.f, 0.f);
  for (; i < n4; i += stride) p[i] = z;
}

// ---- Kernel 4: COO spmm scatter: out[rows[e]] += vals[e]*y[cols[e]] -------
// One wave per edge per iteration; lane covers cols {lane, lane+32}.
// L2-resident (y/out = 38MB each << 192MB L2); prefetch next edge's row.
__global__ __launch_bounds__(256) void spmm_kernel(
    const int* __restrict__ rows, const int* __restrict__ cols,
    const float* __restrict__ vals, const float* __restrict__ y,
    float* __restrict__ out, int E) {
  int lane = threadIdx.x & 31;
  int wave = blockIdx.x * (blockDim.x >> 5) + (threadIdx.x >> 5);
  int nw   = gridDim.x * (blockDim.x >> 5);
  for (int e = wave; e < E; e += nw) {
    int en = e + nw;
    if (en < E) {  // global_prefetch_b8 the next iteration's gather row
      __builtin_prefetch(&y[(size_t)cols[en] * 64 + lane], 0, 0);
    }
    int   c = cols[e];
    int   r = rows[e];
    float v = vals[e];
    float y0 = y[(size_t)c * 64 + lane];
    float y1 = y[(size_t)c * 64 + 32 + lane];
    atomicAdd(&out[(size_t)r * 64 + lane],      v * y0);
    atomicAdd(&out[(size_t)r * 64 + 32 + lane], v * y1);
  }
}

// ---- Kernel 5: x = (relu?)(x); acc += x (b128 traffic) --------------------
__global__ void post_agg_kernel(float4* __restrict__ x, float4* __restrict__ acc,
                                int n4, int do_relu) {
  int i = blockIdx.x * blockDim.x + threadIdx.x;
  int stride = gridDim.x * blockDim.x;
  for (; i < n4; i += stride) {
    float4 v = x[i];
    if (do_relu) {
      v.x = fmaxf(v.x, 0.f); v.y = fmaxf(v.y, 0.f);
      v.z = fmaxf(v.z, 0.f); v.w = fmaxf(v.w, 0.f);
    }
    x[i] = v;
    float4 a = acc[i];
    a.x += v.x; a.y += v.y; a.z += v.z; a.w += v.w;
    acc[i] = a;
  }
}

// ---- Kernel 6: gather + pair features + MLP scorer + clip -----------------
__global__ __launch_bounds__(128) void scorer_kernel(
    const float* __restrict__ acc,
    const int* __restrict__ user_idx, const int* __restrict__ item_idx,
    const float* __restrict__ user_bias, const float* __restrict__ item_bias,
    const float* __restrict__ s1W, const float* __restrict__ s1b,
    const float* __restrict__ s2W, const float* __restrict__ s2b,
    const float* __restrict__ gmean, float* __restrict__ out) {
  __shared__ __bf16 P[16][256];
  __shared__ float  Sbuf[16][SLD];
  int tid = threadIdx.x, lane = tid & 31, w = tid >> 5;
  int q0 = blockIdx.x * 16;
  // build pair = [z_u | z_i | z_u*z_i | |z_u-z_i|], z = acc/3
#pragma unroll
  for (int i = 0; i < 8; ++i) {
    int e = tid + i * 128;
    int q = e >> 6, c = e & 63;
    int u  = user_idx[q0 + q];
    int it = item_idx[q0 + q];
    float zu = acc[(size_t)u * 64 + c] * (1.0f / 3.0f);
    float zi = acc[((size_t)NUSERS + it) * 64 + c] * (1.0f / 3.0f);
    P[q][c]       = (__bf16)zu;
    P[q][64 + c]  = (__bf16)zi;
    P[q][128 + c] = (__bf16)(zu * zi);
    P[q][192 + c] = (__bf16)fabsf(zu - zi);
  }
  __syncthreads();
  wave_gemm_bias_relu<256>(&P[0][0], 256, s1W, 256, s1b,
                           &Sbuf[0][0], SLD, lane, w * 16, true);
  __syncthreads();
  // final 64->1 dot per row (8 threads/row), bias, clip
  int row = tid >> 3, g = tid & 7;
  float d = 0.f;
#pragma unroll
  for (int i = 0; i < 8; ++i) {
    int c = g * 8 + i;
    d += Sbuf[row][c] * s2W[c];
  }
  d += __shfl_xor(d, 4, 8); d += __shfl_xor(d, 2, 8); d += __shfl_xor(d, 1, 8);
  if (g == 0) {
    int u  = user_idx[q0 + row];
    int it = item_idx[q0 + row];
    float pred = gmean[0] + user_bias[u] + item_bias[it] + d + s2b[0];
    pred = fminf(fmaxf(pred, 1.0f), 5.0f);
    out[q0 + row] = pred;
  }
}

// ---------------------------------------------------------------------------
extern "C" void kernel_launch(void* const* d_in, const int* in_sizes, int n_in,
                              void* d_out, int out_size, void* d_ws,
                              size_t ws_size, hipStream_t stream) {
  const float* user_features = (const float*)d_in[0];
  const float* item_features = (const float*)d_in[1];
  const float* adj_vals      = (const float*)d_in[2];
  const float* ueW = (const float*)d_in[3];
  const float* ueB = (const float*)d_in[4];
  const float* ueG = (const float*)d_in[5];
  const float* ueBE= (const float*)d_in[6];
  const float* ieW = (const float*)d_in[7];
  const float* ieB = (const float*)d_in[8];
  const float* ieG = (const float*)d_in[9];
  const float* ieBE= (const float*)d_in[10];
  const float* uid_emb = (const float*)d_in[11];
  const float* iid_emb = (const float*)d_in[12];
  const float* ufW = (const float*)d_in[13];
  const float* ufB = (const float*)d_in[14];
  const float* ufG = (const float*)d_in[15];
  const float* ufBE= (const float*)d_in[16];
  const float* ifW = (const float*)d_in[17];
  const float* ifB = (const float*)d_in[18];
  const float* ifG = (const float*)d_in[19];
  const float* ifBE= (const float*)d_in[20];
  const float* gcnW = (const float*)d_in[21];
  const float* gcnB = (const float*)d_in[22];
  const float* user_bias = (const float*)d_in[23];
  const float* item_bias = (const float*)d_in[24];
  const float* s1W = (const float*)d_in[25];
  const float* s1b = (const float*)d_in[26];
  const float* s2W = (const float*)d_in[27];
  const float* s2b = (const float*)d_in[28];
  const float* gm  = (const float*)d_in[29];
  const int* adj_rows = (const int*)d_in[30];
  const int* adj_cols = (const int*)d_in[31];
  const int* user_idx = (const int*)d_in[32];
  const int* item_idx = (const int*)d_in[33];

  // workspace: x (current layer, also spmm target), y (linear out), acc
  float* xbuf = (float*)d_ws;
  float* ybuf = xbuf + (size_t)NNODES * HDIM;
  float* accb = ybuf + (size_t)NNODES * HDIM;

  int E = in_sizes[2];
  int nElems = NNODES * HDIM;

  enc_fuse_kernel<<<NUSERS / 16, 128, 0, stream>>>(
      user_features, ueW, ueB, ueG, ueBE, uid_emb, ufW, ufB, ufG, ufBE,
      xbuf, accb);
  enc_fuse_kernel<<<NITEMS / 16, 128, 0, stream>>>(
      item_features, ieW, ieB, ieG, ieBE, iid_emb, ifW, ifB, ifG, ifBE,
      xbuf + (size_t)NUSERS * HDIM, accb + (size_t)NUSERS * HDIM);

  for (int l = 0; l < 2; ++l) {
    gcn_linear_kernel<<<NNODES / 16, 128, 0, stream>>>(
        xbuf, gcnW + (size_t)l * HDIM * HDIM, gcnB + (size_t)l * HDIM, ybuf);
    zero_kernel<<<2048, 256, 0, stream>>>((float4*)xbuf, nElems / 4);
    spmm_kernel<<<4096, 256, 0, stream>>>(adj_rows, adj_cols, adj_vals, ybuf,
                                          xbuf, E);
    post_agg_kernel<<<2048, 256, 0, stream>>>((float4*)xbuf, (float4*)accb,
                                              nElems / 4, (l == 0) ? 1 : 0);
  }

  scorer_kernel<<<16384 / 16, 128, 0, stream>>>(
      accb, user_idx, item_idx, user_bias, item_bias, s1W, s1b, s2W, s2b, gm,
      (float*)d_out);
}
